// DDILLM_38860864094521
// MI455X (gfx1250) — matrix operand order repaired and checked by reading, not compiled, not run on hardware
//
#include <hip/hip_runtime.h>
#include <hip/hip_bf16.h>
#include <stdint.h>

#define NND 131072
#define NED 524288
#define BGR 4096

typedef __attribute__((ext_vector_type(16))) _Float16 v16h;
typedef __attribute__((ext_vector_type(8)))  float    v8f;
typedef __attribute__((ext_vector_type(4)))  int      v4i_t;

#if defined(__has_builtin)
#  if __has_builtin(__builtin_amdgcn_global_load_async_to_lds_b128) && \
      __has_builtin(__builtin_amdgcn_s_wait_asynccnt)
#    define USE_ASYNC_LDS 1
#  endif
#endif
#ifndef USE_ASYNC_LDS
#  define USE_ASYNC_LDS 0
#endif

#if USE_ASYNC_LDS
typedef __attribute__((address_space(1))) v4i_t gv4i_t;
typedef __attribute__((address_space(3))) v4i_t sv4i_t;
__device__ __forceinline__ void async_copy16(const void* g, void* l) {
  // GLOBAL_LOAD_ASYNC_TO_LDS_B128: tracked by ASYNCcnt
  __builtin_amdgcn_global_load_async_to_lds_b128(
      (gv4i_t*)(uintptr_t)g, (sv4i_t*)(unsigned int)(uintptr_t)l, 0, 0);
}
#endif

// ---------- WMMA fragment loaders (CDNA5 16-bit layouts, wave32) ----------
// A 16x32 f16 (row stride 64 halfs in LDS tile): lane m=lane&15, hh=lane>>4
__device__ __forceinline__ v16h load_a_frag(const _Float16* As, int lane) {
  int m = lane & 15, hh = lane >> 4;
  const _Float16* p = As + m * 64 + hh * 16;
  v16h a;
#pragma unroll
  for (int t = 0; t < 16; ++t) a[t] = p[t];
  return a;
}
// B 32x16 f16 from LDS tile stored [n][k] (64 k per n): 32 contiguous bytes/lane
__device__ __forceinline__ v16h load_b_frag(const _Float16* Ws, int n0, int ks, int lane) {
  int n = lane & 15, kh = lane >> 4;
  const _Float16* p = Ws + (n0 + n) * 64 + ks + kh * 16;
  v16h b;
#pragma unroll
  for (int t = 0; t < 16; ++t) b[t] = p[t];
  return b;
}

// ---------- pre-conversion kernels ----------
// activations f32 [M,K] -> f16 [M,Kp] zero-padded
__global__ void ddillm_cvt_a(const float* __restrict__ in, _Float16* __restrict__ out,
                             size_t M, int K, int Kp) {
  size_t i = (size_t)blockIdx.x * blockDim.x + threadIdx.x;
  if (i >= M * (size_t)Kp) return;
  int c = (int)(i % Kp);
  size_t r = i / Kp;
  out[i] = (c < K) ? (_Float16)in[r * K + c] : (_Float16)0.f;
}
// weights f32 [K,Nc] -> f16 transposed [Nc,Kp] zero-padded
__global__ void ddillm_cvt_w(const float* __restrict__ in, _Float16* __restrict__ out,
                             int K, int Kp, int Nc) {
  int i = blockIdx.x * blockDim.x + threadIdx.x;
  if (i >= Nc * Kp) return;
  int k = i % Kp, n = i / Kp;
  out[i] = (k < K) ? (_Float16)in[(size_t)k * Nc + n] : (_Float16)0.f;
}

// ---------- WMMA GEMM: C[M,Nc] = act(A[M,Kp](f16) @ Wt[Nc,Kp](f16)^T + bias) ----------
// M % 128 == 0, Nc % 64 == 0, Kp % 64 == 0.
__global__ __launch_bounds__(256)
void ddillm_wmma_gemm(const _Float16* __restrict__ A, int Kp,
                      const _Float16* __restrict__ Wt,
                      const float* __restrict__ bias, float* __restrict__ C,
                      int Nc, int act) {
  __shared__ _Float16 As[128 * 64];   // [row][k]
  __shared__ _Float16 Ws[64 * 64];    // [n][k]
  const int bm = blockIdx.x * 128, bn = blockIdx.y * 64;
  const int tid = threadIdx.x, wave = tid >> 5, lane = tid & 31;
  const int wm = (wave & 3) * 32;   // 0,32,64,96
  const int wn = (wave >> 2) * 32;  // 0,32
  const int cn = lane & 15, chalf = lane >> 4;

  v8f acc[2][2];
#pragma unroll
  for (int j = 0; j < 2; ++j) {
    float b0 = bias ? bias[bn + wn + 16 * j + cn] : 0.f;
#pragma unroll
    for (int i = 0; i < 2; ++i)
#pragma unroll
      for (int r = 0; r < 8; ++r) acc[i][j][r] = b0;
  }

  // staging map: A -> 256 threads x 64B ; W -> 256 threads x 32B (all 16B-aligned)
  const int ar_ = tid >> 1, ac_ = (tid & 1) * 32;  // row, half-col
  const int wn_ = tid >> 2, wc_ = (tid & 3) * 16;  // n, quarter-col

  for (int k0 = 0; k0 < Kp; k0 += 64) {
    const _Float16* ga = A + (size_t)(bm + ar_) * Kp + k0 + ac_;
    const _Float16* gw = Wt + (size_t)(bn + wn_) * Kp + k0 + wc_;
    _Float16* la = &As[ar_ * 64 + ac_];
    _Float16* lw = &Ws[wn_ * 64 + wc_];
    if (k0 + 64 < Kp) __builtin_prefetch(ga + 64, 0, 1);  // global_prefetch_b8
#if USE_ASYNC_LDS
#pragma unroll
    for (int q = 0; q < 4; ++q) async_copy16(ga + 8 * q, la + 8 * q);
#pragma unroll
    for (int q = 0; q < 2; ++q) async_copy16(gw + 8 * q, lw + 8 * q);
    __builtin_amdgcn_s_wait_asynccnt(0);
    __syncthreads();
#else
    {
      const uint4* g4 = (const uint4*)ga; uint4* l4 = (uint4*)la;
#pragma unroll
      for (int q = 0; q < 4; ++q) l4[q] = g4[q];
    }
    {
      const uint4* g4 = (const uint4*)gw; uint4* l4 = (uint4*)lw;
      l4[0] = g4[0]; l4[1] = g4[1];
    }
    __syncthreads();
#endif

#pragma unroll
    for (int ks = 0; ks < 64; ks += 32) {
      v16h bf0 = load_b_frag(Ws, wn + 0, ks, lane);
      v16h bf1 = load_b_frag(Ws, wn + 16, ks, lane);
#pragma unroll
      for (int i = 0; i < 2; ++i) {
        v16h af = load_a_frag(As + (wm + 16 * i) * 64 + ks, lane);
        acc[i][0] = __builtin_amdgcn_wmma_f32_16x16x32_f16(false, af, false, bf0,
                                                           (short)0, acc[i][0], false, false);
        acc[i][1] = __builtin_amdgcn_wmma_f32_16x16x32_f16(false, af, false, bf1,
                                                           (short)0, acc[i][1], false, false);
      }
    }
    __syncthreads();
  }

  // D layout: lane col = cn, VGPR r -> row = r + 8*chalf (per 16x16 subtile)
#pragma unroll
  for (int i = 0; i < 2; ++i)
#pragma unroll
    for (int j = 0; j < 2; ++j) {
      int col = bn + wn + 16 * j + cn;
      size_t rbase = (size_t)(bm + wm + 16 * i + 8 * chalf);
#pragma unroll
      for (int r = 0; r < 8; ++r) {
        float v = acc[i][j][r];
        if (act) v = fmaxf(v, 0.f);
        C[(rbase + r) * Nc + col] = v;
      }
    }
}

// ---------- attention dot products: el/er [N,H] ----------
__global__ void ddillm_attn_dots(const float* __restrict__ h, const float* __restrict__ al,
                                 const float* __restrict__ ar, float* __restrict__ el,
                                 float* __restrict__ er, int Nn, int H, int Dh) {
  int i = blockIdx.x * blockDim.x + threadIdx.x;
  if (i >= Nn * H) return;
  int n = i / H, hd = i % H;
  const float* hp = h + (size_t)n * H * Dh + hd * Dh;
  float sl = 0.f, sr = 0.f;
  for (int d = 0; d < Dh; ++d) { float x = hp[d]; sl += x * al[hd * Dh + d]; sr += x * ar[hd * Dh + d]; }
  el[i] = sl; er[i] = sr;
}

__global__ void ddillm_zero(float* __restrict__ p, size_t n) {
  size_t i = (size_t)blockIdx.x * blockDim.x + threadIdx.x;
  if (i < n) p[i] = 0.f;
}

// per-edge exp(leakyrelu) + denominator accumulation (logits are small: max skip is safe)
__global__ void ddillm_edge_exp(const int* __restrict__ src, const int* __restrict__ dst,
                                const float* __restrict__ el, const float* __restrict__ er,
                                float* __restrict__ earr, float* __restrict__ ssum,
                                int Ecnt, int H) {
  int i = blockIdx.x * blockDim.x + threadIdx.x;
  if (i >= Ecnt * H) return;
  int e = i / H, hd = i % H;
  int s = src[e], d = dst[e];
  float t = el[s * H + hd] + er[d * H + hd];
  t = t > 0.f ? t : 0.2f * t;
  float a = __expf(t);
  earr[i] = a;
  atomicAdd(&ssum[d * H + hd], a);
}

// normalized message scatter-add: one thread per (edge, 4 features)
__global__ void ddillm_edge_agg(const int* __restrict__ src, const int* __restrict__ dst,
                                const float* __restrict__ earr, const float* __restrict__ ssum,
                                const float* __restrict__ h, float* __restrict__ out,
                                int Ecnt, int H, int dhShift, int qpr /* = H*Dh/4 */) {
  size_t i = (size_t)blockIdx.x * blockDim.x + threadIdx.x;
  if (i >= (size_t)Ecnt * qpr) return;
  int e = (int)(i / qpr);
  int q4 = (int)(i % qpr) * 4;       // feature offset within [0, H*Dh)
  int hd = q4 >> dhShift;
  int s = src[e], d = dst[e];
  float alpha = earr[(size_t)e * H + hd] / ssum[d * H + hd];
  int Dout = qpr * 4;
  const float4 hv = *(const float4*)(h + (size_t)s * Dout + q4);
  float* op = out + (size_t)d * Dout + q4;
  atomicAdd(op + 0, alpha * hv.x);
  atomicAdd(op + 1, alpha * hv.y);
  atomicAdd(op + 2, alpha * hv.z);
  atomicAdd(op + 3, alpha * hv.w);
}

__global__ void ddillm_bias_act(float* __restrict__ x, const float* __restrict__ b,
                                size_t total, int D, int act) {
  size_t i = (size_t)blockIdx.x * blockDim.x + threadIdx.x;
  if (i >= total) return;
  float v = x[i] + b[i % D];
  if (act) v = fmaxf(v, 0.f);
  x[i] = v;
}

// mean over 32 contiguous nodes per graph (gid = arange(N)//32) -> pair[:, colOff:colOff+D]
__global__ void ddillm_pool(const float* __restrict__ h, float* __restrict__ pair,
                            int Bn, int D, int colOff, int stride) {
  int i = blockIdx.x * blockDim.x + threadIdx.x;
  if (i >= Bn * D) return;
  int b = i / D, c = i % D;
  const float* p = h + (size_t)b * 32 * D + c;
  float s = 0.f;
#pragma unroll
  for (int r = 0; r < 32; ++r) s += p[(size_t)r * D];
  pair[(size_t)b * stride + colOff + c] = s * (1.f / 32.f);
}

__global__ void ddillm_add(const float* __restrict__ a, const float* __restrict__ b,
                           float* __restrict__ y, size_t n) {
  size_t i = (size_t)blockIdx.x * blockDim.x + threadIdx.x;
  if (i < n) y[i] = a[i] + b[i];
}

// wave32 LayerNorm: one wave per row of D=512
__global__ __launch_bounds__(256)
void ddillm_layernorm(float* __restrict__ x, const float* __restrict__ gamma,
                      const float* __restrict__ beta, int Bn, int D) {
  int row = (blockIdx.x * blockDim.x + threadIdx.x) >> 5;
  int lane = threadIdx.x & 31;
  if (row >= Bn) return;
  float* xp = x + (size_t)row * D;
  float s = 0.f, s2 = 0.f;
  for (int c = lane; c < D; c += 32) { float v = xp[c]; s += v; s2 += v * v; }
#pragma unroll
  for (int o = 16; o > 0; o >>= 1) { s += __shfl_xor(s, o, 32); s2 += __shfl_xor(s2, o, 32); }
  float mu = s / (float)D;
  float var = s2 / (float)D - mu * mu;
  float inv = rsqrtf(var + 1e-5f);
  for (int c = lane; c < D; c += 32) xp[c] = (xp[c] - mu) * inv * gamma[c] + beta[c];
}

// final 256 -> 1 matvec + sigmoid: one wave per row
__global__ __launch_bounds__(256)
void ddillm_head(const float* __restrict__ c1, const float* __restrict__ Wc2,
                 const float* __restrict__ bc2, float* __restrict__ out, int Bn, int D) {
  int row = (blockIdx.x * blockDim.x + threadIdx.x) >> 5;
  int lane = threadIdx.x & 31;
  if (row >= Bn) return;
  const float* p = c1 + (size_t)row * D;
  float s = 0.f;
  for (int c = lane; c < D; c += 32) s += p[c] * Wc2[c];
#pragma unroll
  for (int o = 16; o > 0; o >>= 1) s += __shfl_xor(s, o, 32);
  if (lane == 0) out[row] = 1.f / (1.f + __expf(-(s + bc2[0])));
}

extern "C" void kernel_launch(void* const* d_in, const int* in_sizes, int n_in,
                              void* d_out, int out_size, void* d_ws, size_t ws_size,
                              hipStream_t stream) {
  (void)in_sizes; (void)n_in; (void)out_size; (void)ws_size;
  const float* feat[2] = {(const float*)d_in[0], (const float*)d_in[1]};
  const float* llm = (const float*)d_in[2];
  const int* srcs[2] = {(const int*)d_in[3], (const int*)d_in[5]};
  const int* dsts[2] = {(const int*)d_in[4], (const int*)d_in[6]};
  const float* Wl[3] = {(const float*)d_in[8],  (const float*)d_in[12], (const float*)d_in[16]};
  const float* al[3] = {(const float*)d_in[9],  (const float*)d_in[13], (const float*)d_in[17]};
  const float* ar[3] = {(const float*)d_in[10], (const float*)d_in[14], (const float*)d_in[18]};
  const float* bl[3] = {(const float*)d_in[11], (const float*)d_in[15], (const float*)d_in[19]};
  const float* Wpg = (const float*)d_in[20]; const float* bpg = (const float*)d_in[21];
  const float* Wpl = (const float*)d_in[22]; const float* bpl = (const float*)d_in[23];
  const float* Wv  = (const float*)d_in[24]; const float* bv  = (const float*)d_in[25];
  const float* Wo  = (const float*)d_in[26]; const float* bo  = (const float*)d_in[27];
  const float* gamma = (const float*)d_in[28]; const float* beta = (const float*)d_in[29];
  const float* Wc1 = (const float*)d_in[30]; const float* bc1 = (const float*)d_in[31];
  const float* Wc2 = (const float*)d_in[32]; const float* bc2 = (const float*)d_in[33];

  char* wsb = (char*)d_ws;
  size_t off = 0;
  auto carve = [&](size_t bytes) -> void* {
    void* p = wsb + off;
    off = (off + bytes + 255) & ~(size_t)255;
    return p;
  };
  float*    t0   = (float*)carve((size_t)NND * 256 * 4);
  float*    t1   = (float*)carve((size_t)NND * 256 * 4);
  _Float16* ah   = (_Float16*)carve((size_t)NND * 128 * 2);  // padded f16 activations
  _Float16* wh   = (_Float16*)carve((size_t)1024 * 512 * 2); // padded f16 transposed weights
  float*    el   = (float*)carve((size_t)NND * 4 * 4);
  float*    er   = (float*)carve((size_t)NND * 4 * 4);
  float*    earr = (float*)carve((size_t)NED * 4 * 4);
  float*    ssum = (float*)carve((size_t)NND * 4 * 4);
  float*    pair = (float*)carve((size_t)BGR * 512 * 4);
  float*    gp   = (float*)carve((size_t)BGR * 512 * 4);
  float*    lp   = (float*)carve((size_t)BGR * 512 * 4);
  float*    vb   = (float*)carve((size_t)BGR * 512 * 4);
  float*    atb  = (float*)carve((size_t)BGR * 512 * 4);
  float*    xb   = (float*)carve((size_t)BGR * 512 * 4);
  float*    c1   = (float*)carve((size_t)BGR * 256 * 4);

  auto run_gemm = [&](const float* Araw, int K, const float* Wraw, const float* bias,
                      float* Cout, size_t M, int Nc, int act) {
    int Kp = (K + 63) & ~63;
    ddillm_cvt_a<<<(unsigned)((M * Kp + 255) / 256), 256, 0, stream>>>(Araw, ah, M, K, Kp);
    ddillm_cvt_w<<<((size_t)Nc * Kp + 255) / 256, 256, 0, stream>>>(Wraw, wh, K, Kp, Nc);
    ddillm_wmma_gemm<<<dim3((unsigned)(M / 128), Nc / 64), 256, 0, stream>>>(
        ah, Kp, wh, bias, Cout, Nc, act);
  };

  const int Din[3]  = {6, 128, 128};
  const int Dout[3] = {128, 128, 256};
  const int DhS[3]  = {5, 5, 6};   // log2(Dh)
  const int rel[3]  = {1, 1, 0};
  const int H = 4;

  for (int side = 0; side < 2; ++side) {
    const int* src = srcs[side];
    const int* dst = dsts[side];
    for (int l = 0; l < 3; ++l) {
      const float* in = (l == 0) ? feat[side] : t1;
      int Dh = 1 << DhS[l], qpr = (H * Dh) >> 2;
      run_gemm(in, Din[l], Wl[l], nullptr, t0, NND, Dout[l], 0);
      ddillm_attn_dots<<<(NND * H + 255) / 256, 256, 0, stream>>>(
          t0, al[l], ar[l], el, er, NND, H, Dh);
      ddillm_zero<<<(NND * H + 255) / 256, 256, 0, stream>>>(ssum, (size_t)NND * H);
      ddillm_zero<<<((size_t)NND * Dout[l] + 255) / 256, 256, 0, stream>>>(t1, (size_t)NND * Dout[l]);
      ddillm_edge_exp<<<(NED * H + 255) / 256, 256, 0, stream>>>(
          src, dst, el, er, earr, ssum, NED, H);
      ddillm_edge_agg<<<((size_t)NED * qpr + 255) / 256, 256, 0, stream>>>(
          src, dst, earr, ssum, t0, t1, NED, H, DhS[l], qpr);
      ddillm_bias_act<<<((size_t)NND * Dout[l] + 255) / 256, 256, 0, stream>>>(
          t1, bl[l], (size_t)NND * Dout[l], Dout[l], rel[l]);
    }
    ddillm_pool<<<(BGR * 256 + 255) / 256, 256, 0, stream>>>(t1, pair, BGR, 256, side * 256, 512);
  }

  // fusion: gp = pair@Wpg+bpg ; lp = llm@Wpl+bpl ; attn = (gp@Wv+bv)@Wo+bo ; x = attn+lp ; LN
  run_gemm(pair, 512,  Wpg, bpg, gp,  BGR, 512, 0);
  run_gemm(llm,  1024, Wpl, bpl, lp,  BGR, 512, 0);
  run_gemm(gp,   512,  Wv,  bv,  vb,  BGR, 512, 0);
  run_gemm(vb,   512,  Wo,  bo,  atb, BGR, 512, 0);
  ddillm_add<<<((size_t)BGR * 512 + 255) / 256, 256, 0, stream>>>(atb, lp, xb, (size_t)BGR * 512);
  ddillm_layernorm<<<(BGR * 32 + 255) / 256, 256, 0, stream>>>(xb, gamma, beta, BGR, 512);
  run_gemm(xb, 512, Wc1, bc1, c1, BGR, 256, 1);
  ddillm_head<<<(BGR * 32 + 255) / 256, 256, 0, stream>>>(c1, Wc2, bc2, (float*)d_out, BGR, 256);
}
